// RumorDetect_73607149519600
// MI455X (gfx1250) — matrix operand order repaired and checked by reading, not compiled, not run on hardware
//
#include <hip/hip_runtime.h>

#define TT 8
#define NN 4096
#define NFEATC 64
#define NHIDC 128
#define NCLASSC 64
#define NCC 4
#define DDC 128
#define H1S 2048
#define H2OUT 8192
#define EPSF 1e-5f

typedef __attribute__((ext_vector_type(16))) _Float16 v16h;
typedef __attribute__((ext_vector_type(8)))  _Float16 v8h;
typedef __attribute__((ext_vector_type(8)))  float    v8f;

__device__ __forceinline__ v16h cat8(v8h lo, v8h hi) {
  return __builtin_shufflevector(lo, hi, 0,1,2,3,4,5,6,7,8,9,10,11,12,13,14,15);
}

// ---------------------------------------------------------------------------
// Generic batched GEMM: C[t] = act(A[t] (MxK,f16) @ B[t] (KxN,f16) + bias)
// Block tile 128x64, 8 waves, each wave a 32x32 patch via 4x v_wmma 16x16x32.
// Double-buffered LDS: next tile's global loads are issued before the current
// tile's WMMAs so load latency overlaps matrix ops.
// M == 4096 always here; N in {64,128}; K % 32 == 0.
// ---------------------------------------------------------------------------
template<int CF16, int HASB, int RELU>
__global__ __launch_bounds__(256)
void gemm_f16_wmma(const _Float16* __restrict__ A, long strideA, int lda,
                   const _Float16* __restrict__ B, long strideB, int ldb,
                   void* __restrict__ Cv, long strideC, int ldc,
                   const float* __restrict__ bias, int K)
{
  const int t = blockIdx.z;
  const _Float16* Ab = A + (long)t * strideA;
  const _Float16* Bb = B + (long)t * strideB;
  const int rowBlk = blockIdx.x * 128;
  const int colBlk = blockIdx.y * 64;

  __shared__ __align__(16) _Float16 sA[2][128 * 40];   // row-major, stride 40
  __shared__ __align__(16) _Float16 sB[2][64 * 40];    // transposed: [n][k]

  const int tid  = threadIdx.x;
  const int lane = tid & 31;
  const int w    = tid >> 5;
  const int wm   = w & 3;        // 0..3 -> 32-row group
  const int wn   = w >> 2;       // 0..1 -> 32-col group
  const int rsel = lane & 15;
  const int khalf = lane >> 4;

  const int ar = tid >> 1, ac = (tid & 1) * 16;        // A stage mapping
  const int bk = tid >> 3, bn0 = (tid & 7) * 8;        // B stage mapping

  v8f acc[2][2] = {};
  v8h ra0, ra1, rb0;

  // prologue: tile 0 -> regs -> LDS buf 0
  {
    const _Float16* asrc = Ab + (long)(rowBlk + ar) * lda + ac;
    ra0 = *(const v8h*)(asrc);
    ra1 = *(const v8h*)(asrc + 8);
    rb0 = *(const v8h*)(Bb + (long)bk * ldb + colBlk + bn0);
    *(v8h*)&sA[0][ar * 40 + ac]     = ra0;
    *(v8h*)&sA[0][ar * 40 + ac + 8] = ra1;
    #pragma unroll
    for (int i = 0; i < 8; ++i) sB[0][(bn0 + i) * 40 + bk] = rb0[i];
  }

  int cur = 0;
  for (int k0 = 0; k0 < K; k0 += 32) {
    __syncthreads();                                   // buf[cur] staged
    const bool hasNext = (k0 + 32) < K;
    if (hasNext) {                                     // issue next tile loads
      const _Float16* asrc = Ab + (long)(rowBlk + ar) * lda + k0 + 32 + ac;
      ra0 = *(const v8h*)(asrc);
      ra1 = *(const v8h*)(asrc + 8);
      rb0 = *(const v8h*)(Bb + (long)(k0 + 32 + bk) * ldb + colBlk + bn0);
    }

    // compute from buf[cur] while the loads above are in flight
    v16h afr[2], bfr[2];
    #pragma unroll
    for (int i = 0; i < 2; ++i) {
      const _Float16* ap = &sA[cur][(wm * 32 + i * 16 + rsel) * 40];
      v8h lo = *(const v8h*)(ap + khalf * 8);
      v8h hi = *(const v8h*)(ap + 16 + khalf * 8);
      afr[i] = cat8(lo, hi);
    }
    #pragma unroll
    for (int j = 0; j < 2; ++j) {
      const _Float16* bp = &sB[cur][(wn * 32 + j * 16 + rsel) * 40 + khalf * 8];
      v8h lo = *(const v8h*)(bp);
      v8h hi = *(const v8h*)(bp + 16);
      bfr[j] = cat8(lo, hi);
    }
    #pragma unroll
    for (int i = 0; i < 2; ++i)
      #pragma unroll
      for (int j = 0; j < 2; ++j)
        acc[i][j] = __builtin_amdgcn_wmma_f32_16x16x32_f16(
            false, afr[i], false, bfr[j], (short)0, acc[i][j], false, false);

    __syncthreads();                                   // everyone done with buf[cur^1]
    if (hasNext) {
      *(v8h*)&sA[cur ^ 1][ar * 40 + ac]     = ra0;
      *(v8h*)&sA[cur ^ 1][ar * 40 + ac + 8] = ra1;
      #pragma unroll
      for (int i = 0; i < 8; ++i) sB[cur ^ 1][(bn0 + i) * 40 + bk] = rb0[i];
    }
    cur ^= 1;
  }

  // epilogue: C layout = VGPR v -> M = v (+8 if lane>=16), N = lane&15
  _Float16* Ch = (_Float16*)Cv + (long)t * strideC;
  float*    Cf = (float*)Cv + (long)t * strideC;
  #pragma unroll
  for (int i = 0; i < 2; ++i) {
    #pragma unroll
    for (int j = 0; j < 2; ++j) {
      int colg = colBlk + wn * 32 + j * 16 + rsel;
      float bv = HASB ? bias[colg] : 0.0f;
      #pragma unroll
      for (int v = 0; v < 8; ++v) {
        int rowg = rowBlk + wm * 32 + i * 16 + v + (khalf << 3);
        float x = acc[i][j][v] + bv;
        if (RELU) x = fmaxf(x, 0.0f);
        if (CF16) Ch[(long)rowg * ldc + colg] = (_Float16)x;
        else      Cf[(long)rowg * ldc + colg] = x;
      }
    }
  }
}

// ---------------------------------------------------------------------------
// Fused gram kernel: out1[t][n] = sum_m relu(ne[n] . ne[m]) * state[m]
// Block = 128 n-rows (8 waves x 16), loop over m in 16-tiles, K=128 (4 WMMAs).
// Never materializes the 4096x4096 gram matrix. Double-buffered m staging.
// B fragments preloaded into distinct registers so all ds_loads can be in
// flight while the WMMA accumulator chain executes.
// ---------------------------------------------------------------------------
__global__ __launch_bounds__(256)
void gram_state_wmma(const _Float16* __restrict__ neh,
                     const float* __restrict__ state,
                     float* __restrict__ out1)
{
  const int t = blockIdx.y;
  const _Float16* ne = neh + (long)t * NN * DDC;
  const int nBase = blockIdx.x * 128;
  const int tid = threadIdx.x, lane = tid & 31, w = tid >> 5;
  const int rsel = lane & 15, khalf = lane >> 4;

  // preload A fragments: this wave's 16 rows x K=128 (4 chunks of 32)
  v16h afr[4];
  {
    const _Float16* arow = ne + (long)(nBase + w * 16 + rsel) * DDC;
    #pragma unroll
    for (int kc = 0; kc < 4; ++kc) {
      v8h lo = *(const v8h*)(arow + kc * 32 + khalf * 8);
      v8h hi = *(const v8h*)(arow + kc * 32 + 16 + khalf * 8);
      afr[kc] = cat8(lo, hi);
    }
  }

  __shared__ __align__(16) _Float16 sM[2][16 * 136];   // 16 m-rows x 128, stride 136
  float sums[8] = {0.f,0.f,0.f,0.f,0.f,0.f,0.f,0.f};

  const int mr = tid >> 4, mc = (tid & 15) * 8;        // stage mapping
  v8h rm = *(const v8h*)(ne + (long)mr * DDC + mc);    // tile 0
  *(v8h*)&sM[0][mr * 136 + mc] = rm;

  int cur = 0;
  for (int mt = 0; mt < NN / 16; ++mt) {
    __syncthreads();                                   // buf[cur] staged
    const bool hasNext = (mt + 1) < (NN / 16);
    if (hasNext)
      rm = *(const v8h*)(ne + (long)((mt + 1) * 16 + mr) * DDC + mc);

    // preload all 4 B fragments (distinct registers -> overlapping ds_loads)
    v16h bfr[4];
    #pragma unroll
    for (int kc = 0; kc < 4; ++kc) {
      const _Float16* bp = &sM[cur][rsel * 136 + kc * 32 + khalf * 8];
      v8h lo = *(const v8h*)bp;
      v8h hi = *(const v8h*)(bp + 16);
      bfr[kc] = cat8(lo, hi);
    }
    v8f acc = {};
    #pragma unroll
    for (int kc = 0; kc < 4; ++kc)
      acc = __builtin_amdgcn_wmma_f32_16x16x32_f16(
          false, afr[kc], false, bfr[kc], (short)0, acc, false, false);

    float sv = state[t * NN + mt * 16 + rsel];
    #pragma unroll
    for (int v = 0; v < 8; ++v) sums[v] += fmaxf(acc[v], 0.0f) * sv;

    __syncthreads();
    if (hasNext) *(v8h*)&sM[cur ^ 1][mr * 136 + mc] = rm;
    cur ^= 1;
  }

  // reduce over the 16 lanes holding the same row set
  #pragma unroll
  for (int v = 0; v < 8; ++v) {
    float s = sums[v];
    #pragma unroll
    for (int off = 1; off < 16; off <<= 1) s += __shfl_xor(s, off, 32);
    sums[v] = s;
  }
  if (rsel == 0) {                       // lanes 0 (M=0..7) and 16 (M=8..15)
    int rbase = nBase + w * 16 + khalf * 8;
    #pragma unroll
    for (int v = 0; v < 8; ++v) out1[t * NN + rbase + v] = sums[v];
  }
}

// ---------------------------------------------------------------------------
// Small helper kernels
// ---------------------------------------------------------------------------
__global__ void f2h_kernel(const float* __restrict__ s, _Float16* __restrict__ d, long n) {
  long i = (long)blockIdx.x * 256 + threadIdx.x;
  if (i < n) d[i] = (_Float16)s[i];
}

__global__ void degree_kernel(const float* __restrict__ adj, float* __restrict__ dinv) {
  int row = blockIdx.x;
  __shared__ float red[256];
  float s = 0.f;
  for (int j = threadIdx.x; j < NN; j += 256)
    s += (adj[(long)row * NN + j] > 0.0f) ? 1.0f : 0.0f;
  red[threadIdx.x] = s; __syncthreads();
  for (int st = 128; st > 0; st >>= 1) {
    if (threadIdx.x < st) red[threadIdx.x] += red[threadIdx.x + st];
    __syncthreads();
  }
  if (threadIdx.x == 0) dinv[row] = rsqrtf(red[0] + 1.0f);
}

__global__ void nm_kernel(const float* __restrict__ adj, const float* __restrict__ dinv,
                          _Float16* __restrict__ Nm) {
  long idx = (long)blockIdx.x * 256 + threadIdx.x;
  int i = (int)(idx >> 12), j = (int)(idx & (NN - 1));
  float a = (adj[idx] > 0.f ? 1.f : 0.f) + (i == j ? 1.f : 0.f);
  Nm[idx] = (_Float16)(dinv[i] * a * dinv[j]);
}

// BN over nodes for ne = concat([x_t, H2_t], axis=1); writes f16 ne
__global__ void bn_ne_kernel(const float* __restrict__ x, const float* __restrict__ Hf,
                             const float* __restrict__ g, const float* __restrict__ beta,
                             _Float16* __restrict__ neh) {
  int col = blockIdx.x, t = blockIdx.y;
  const float* src = (col < NFEATC) ? (x + (long)t * NN * NFEATC + col)
                                    : (Hf + (long)t * NN * NFEATC + (col - NFEATC));
  __shared__ float r1[256], r2[256];
  __shared__ float scS, shS;
  float s = 0.f, s2 = 0.f;
  for (int n = threadIdx.x; n < NN; n += 256) {
    float v = src[(long)n * NFEATC]; s += v; s2 += v * v;
  }
  r1[threadIdx.x] = s; r2[threadIdx.x] = s2; __syncthreads();
  for (int st = 128; st > 0; st >>= 1) {
    if (threadIdx.x < st) { r1[threadIdx.x] += r1[threadIdx.x + st]; r2[threadIdx.x] += r2[threadIdx.x + st]; }
    __syncthreads();
  }
  if (threadIdx.x == 0) {
    float m = r1[0] / (float)NN;
    float var = r2[0] / (float)NN - m * m;
    float sc = rsqrtf(var + EPSF) * g[col];
    scS = sc; shS = beta[col] - m * sc;
  }
  __syncthreads();
  float sc = scS, sh = shS;
  for (int n = threadIdx.x; n < NN; n += 256)
    neh[((long)t * NN + n) * DDC + col] = (_Float16)(src[(long)n * NFEATC] * sc + sh);
}

__global__ void logits_kernel(const float* __restrict__ HC, const float* __restrict__ Wm,
                              const float* __restrict__ bm, float* __restrict__ L) {
  long i = (long)blockIdx.x * 256 + threadIdx.x;         // over TT*NN
  const float* h = HC + i * 64;
  #pragma unroll
  for (int c = 0; c < NCC; ++c) {
    float a = bm[c];
    for (int k = 0; k < 64; ++k) a += h[k] * Wm[k * NCC + c];
    L[i * NCC + c] = fmaxf(a, 0.0f);
  }
}

__global__ void lstats_kernel(const float* __restrict__ L, float* __restrict__ Lmean,
                              float* __restrict__ Lrstd) {
  int c = blockIdx.x, t = blockIdx.y;
  __shared__ float r1[256], r2[256];
  float s = 0.f, s2 = 0.f;
  for (int n = threadIdx.x; n < NN; n += 256) {
    float v = L[((long)t * NN + n) * NCC + c]; s += v; s2 += v * v;
  }
  r1[threadIdx.x] = s; r2[threadIdx.x] = s2; __syncthreads();
  for (int st = 128; st > 0; st >>= 1) {
    if (threadIdx.x < st) { r1[threadIdx.x] += r1[threadIdx.x + st]; r2[threadIdx.x] += r2[threadIdx.x + st]; }
    __syncthreads();
  }
  if (threadIdx.x == 0) {
    float m = r1[0] / (float)NN;
    float var = r2[0] / (float)NN - m * m;
    Lmean[t * NCC + c] = m;
    Lrstd[t * NCC + c] = rsqrtf(var + EPSF);
  }
}

__global__ void ids_ss_kernel(const float* __restrict__ L, const float* __restrict__ Lmean,
                              const float* __restrict__ Lrstd, const float* __restrict__ g_cn,
                              const float* __restrict__ b_cn, const _Float16* __restrict__ neh,
                              const float* __restrict__ Wce, const float* __restrict__ bce,
                              int* __restrict__ ids, float* __restrict__ ss) {
  long i = (long)blockIdx.x * 256 + threadIdx.x;
  int t = (int)(i >> 12);
  float best = -3.4e38f; int bi = 0;
  #pragma unroll
  for (int c = 0; c < NCC; ++c) {
    float lv = (L[i * NCC + c] - Lmean[t * NCC + c]) * Lrstd[t * NCC + c] * g_cn[c] + b_cn[c];
    if (lv > best) { best = lv; bi = c; }
  }
  ids[i] = bi;
  float a = bce[0];
  const _Float16* nh = neh + i * DDC;
  for (int k = 0; k < DDC; ++k) a += (float)nh[k] * Wce[k];
  ss[i] = 1.0f / (1.0f + expf(-a));
}

__global__ void class_sum_kernel(const int* __restrict__ ids, const float* __restrict__ ss,
                                 float* __restrict__ ssum, int* __restrict__ cnt) {
  int c = blockIdx.x, t = blockIdx.y;
  __shared__ float rs[256]; __shared__ int rc[256];
  float s = 0.f; int n = 0;
  for (int i = threadIdx.x; i < NN; i += 256) {
    if (ids[t * NN + i] == c) { s += ss[t * NN + i]; ++n; }
  }
  rs[threadIdx.x] = s; rc[threadIdx.x] = n; __syncthreads();
  for (int st = 128; st > 0; st >>= 1) {
    if (threadIdx.x < st) { rs[threadIdx.x] += rs[threadIdx.x + st]; rc[threadIdx.x] += rc[threadIdx.x + st]; }
    __syncthreads();
  }
  if (threadIdx.x == 0) { ssum[t * NCC + c] = rs[0]; cnt[t * NCC + c] = rc[0]; }
}

__global__ void score_top_kernel(const float* __restrict__ ssum, const int* __restrict__ cnt,
                                 float* __restrict__ Cs_out, float* __restrict__ cid_out,
                                 int* __restrict__ idx_ws) {
  int t = blockIdx.x;
  if (threadIdx.x != 0) return;
  float sc[NCC];
  #pragma unroll
  for (int c = 0; c < NCC; ++c)
    sc[c] = ssum[t * NCC + c] / fmaxf((float)cnt[t * NCC + c], 1.0f);
  int i0 = 0;
  for (int c = 1; c < NCC; ++c) if (sc[c] > sc[i0]) i0 = c;
  int i1 = -1;
  for (int c = 0; c < NCC; ++c) {
    if (c == i0) continue;
    if (i1 < 0 || sc[c] > sc[i1]) i1 = c;
  }
  #pragma unroll
  for (int c = 0; c < NCC; ++c) Cs_out[t * NCC + c] = sc[c];
  cid_out[t * 2] = (float)i0; cid_out[t * 2 + 1] = (float)i1;
  idx_ws[t * 2] = i0; idx_ws[t * 2 + 1] = i1;
}

__global__ void mask_kernel(const int* __restrict__ ids, const int* __restrict__ idx_ws,
                            float* __restrict__ tmask) {
  long i = (long)blockIdx.x * 256 + threadIdx.x;
  int t = (int)(i >> 12);
  int id = ids[i];
  tmask[i] = (id == idx_ws[t * 2] ? 1.0f : 0.0f) + (id == idx_ws[t * 2 + 1] ? 1.0f : 0.0f);
}

__global__ void mlp1_kernel(const float* __restrict__ o1, const float* __restrict__ Wm1,
                            const float* __restrict__ bm1, float* __restrict__ o2) {
  int j = blockIdx.x * 256 + threadIdx.x, t = blockIdx.y;
  float acc = bm1[j];
  for (int k = 0; k < NN; ++k) acc += o1[t * NN + k] * Wm1[(long)k * H1S + j];
  o2[t * H1S + j] = fmaxf(acc, 0.0f);
}

__global__ void mlp2_kernel(const float* __restrict__ o2, const float* __restrict__ Wm2,
                            const float* __restrict__ bm2, float* __restrict__ outp) {
  int j = blockIdx.x * 256 + threadIdx.x, t = blockIdx.y;
  float acc = bm2[j];
  for (int k = 0; k < H1S; ++k) acc += o2[t * H1S + k] * Wm2[(long)k * H2OUT + j];
  outp[(long)t * H2OUT + j] = acc;
}

// ---------------------------------------------------------------------------
extern "C" void kernel_launch(void* const* d_in, const int* in_sizes, int n_in,
                              void* d_out, int out_size, void* d_ws, size_t ws_size,
                              hipStream_t stream) {
  const float* x     = (const float*)d_in[0];
  const float* state = (const float*)d_in[1];
  const float* adj   = (const float*)d_in[2];
  const float* W1    = (const float*)d_in[3];
  const float* b1    = (const float*)d_in[4];
  const float* W2    = (const float*)d_in[5];
  const float* b2    = (const float*)d_in[6];
  const float* g_nm  = (const float*)d_in[7];
  const float* be_nm = (const float*)d_in[8];
  const float* Wc    = (const float*)d_in[9];
  const float* bc    = (const float*)d_in[10];
  const float* Wm    = (const float*)d_in[11];
  const float* bm    = (const float*)d_in[12];
  const float* g_cn  = (const float*)d_in[13];
  const float* b_cn  = (const float*)d_in[14];
  const float* Wce   = (const float*)d_in[15];
  const float* bce   = (const float*)d_in[16];
  const float* Wm1   = (const float*)d_in[17];
  const float* bm1   = (const float*)d_in[18];
  const float* Wm2   = (const float*)d_in[19];
  const float* bm2   = (const float*)d_in[20];

  char* ws = (char*)d_ws;
  size_t off = 0;
  auto alloc = [&](size_t bytes) -> void* {
    void* p = ws + off;
    off = (off + bytes + 255) & ~(size_t)255;
    return p;
  };

  float*     dinv = (float*)alloc(NN * 4);
  _Float16*  Nm_h = (_Float16*)alloc((size_t)NN * NN * 2);
  _Float16*  x_h  = (_Float16*)alloc((size_t)TT * NN * NFEATC * 2);
  _Float16*  W1_h = (_Float16*)alloc(NFEATC * NHIDC * 2);
  _Float16*  W2_h = (_Float16*)alloc(NHIDC * NCLASSC * 2);
  _Float16*  Wc_h = (_Float16*)alloc(DDC * 64 * 2);
  _Float16*  X1_h = (_Float16*)alloc((size_t)TT * NN * NHIDC * 2);
  _Float16*  H1_h = (_Float16*)alloc((size_t)TT * NN * NHIDC * 2);
  _Float16*  X2_h = (_Float16*)alloc((size_t)TT * NN * 64 * 2);
  float*     H2_f = (float*)alloc((size_t)TT * NN * 64 * 4);
  _Float16*  ne_h = (_Float16*)alloc((size_t)TT * NN * DDC * 2);
  _Float16*  XC_h = (_Float16*)alloc((size_t)TT * NN * 64 * 2);
  float*     HC_f = (float*)alloc((size_t)TT * NN * 64 * 4);
  float*     Lbuf = (float*)alloc((size_t)TT * NN * NCC * 4);
  float*     Lmean = (float*)alloc(TT * NCC * 4);
  float*     Lrstd = (float*)alloc(TT * NCC * 4);
  int*       ids  = (int*)alloc((size_t)TT * NN * 4);
  float*     ssb  = (float*)alloc((size_t)TT * NN * 4);
  float*     ssum = (float*)alloc(TT * NCC * 4);
  int*       cntb = (int*)alloc(TT * NCC * 4);
  int*       idxt = (int*)alloc(TT * 2 * 4);
  float*     out1 = (float*)alloc((size_t)TT * NN * 4);
  float*     out2 = (float*)alloc((size_t)TT * H1S * 4);
  (void)ws_size; (void)in_sizes; (void)n_in; (void)out_size;

  float* Cs_out    = (float*)d_out;               // (8,4,1)
  float* Tmask_out = (float*)d_out + 32;          // (8,4096,1)
  float* cid_out   = (float*)d_out + 32800;       // (8,2)
  float* final_out = (float*)d_out + 32816;       // (8,4096,2)

  // --- conversions ---
  {
    long n = (long)TT * NN * NFEATC;
    hipLaunchKernelGGL(f2h_kernel, dim3((unsigned)((n + 255) / 256)), dim3(256), 0, stream, x, x_h, n);
    hipLaunchKernelGGL(f2h_kernel, dim3(32), dim3(256), 0, stream, W1, W1_h, (long)NFEATC * NHIDC);
    hipLaunchKernelGGL(f2h_kernel, dim3(32), dim3(256), 0, stream, W2, W2_h, (long)NHIDC * NCLASSC);
    hipLaunchKernelGGL(f2h_kernel, dim3(32), dim3(256), 0, stream, Wc, Wc_h, (long)DDC * 64);
  }
  // --- normalized adjacency (f16) ---
  hipLaunchKernelGGL(degree_kernel, dim3(NN), dim3(256), 0, stream, adj, dinv);
  hipLaunchKernelGGL(nm_kernel, dim3((unsigned)(((size_t)NN * NN) / 256)), dim3(256), 0, stream, adj, dinv, Nm_h);

  // --- GCN layers (WMMA GEMMs) ---
  // X1 = x @ W1                         (f16 out, no bias, no relu)
  gemm_f16_wmma<1,0,0><<<dim3(NN/128, NHIDC/64, TT), dim3(256), 0, stream>>>(
      x_h, (long)NN * NFEATC, NFEATC, W1_h, 0, NHIDC,
      X1_h, (long)NN * NHIDC, NHIDC, nullptr, NFEATC);
  // H1 = relu(Nm @ X1 + b1)             (f16 out, bias, relu)
  gemm_f16_wmma<1,1,1><<<dim3(NN/128, NHIDC/64, TT), dim3(256), 0, stream>>>(
      Nm_h, 0, NN, X1_h, (long)NN * NHIDC, NHIDC,
      H1_h, (long)NN * NHIDC, NHIDC, b1, NN);
  // X2 = H1 @ W2
  gemm_f16_wmma<1,0,0><<<dim3(NN/128, 1, TT), dim3(256), 0, stream>>>(
      H1_h, (long)NN * NHIDC, NHIDC, W2_h, 0, 64,
      X2_h, (long)NN * 64, 64, nullptr, NHIDC);
  // H2 = relu(Nm @ X2 + b2)             (f32 out)
  gemm_f16_wmma<0,1,1><<<dim3(NN/128, 1, TT), dim3(256), 0, stream>>>(
      Nm_h, 0, NN, X2_h, (long)NN * 64, 64,
      H2_f, (long)NN * 64, 64, b2, NN);

  // --- batch-norm of concat([x, h]) -> ne (f16) ---
  hipLaunchKernelGGL(bn_ne_kernel, dim3(DDC, TT), dim3(256), 0, stream, x, H2_f, g_nm, be_nm, ne_h);

  // --- cluster head ---
  gemm_f16_wmma<1,0,0><<<dim3(NN/128, 1, TT), dim3(256), 0, stream>>>(
      ne_h, (long)NN * DDC, DDC, Wc_h, 0, 64,
      XC_h, (long)NN * 64, 64, nullptr, DDC);
  gemm_f16_wmma<0,1,1><<<dim3(NN/128, 1, TT), dim3(256), 0, stream>>>(
      Nm_h, 0, NN, XC_h, (long)NN * 64, 64,
      HC_f, (long)NN * 64, 64, bc, NN);

  hipLaunchKernelGGL(logits_kernel, dim3(TT * NN / 256), dim3(256), 0, stream, HC_f, Wm, bm, Lbuf);
  hipLaunchKernelGGL(lstats_kernel, dim3(NCC, TT), dim3(256), 0, stream, Lbuf, Lmean, Lrstd);
  hipLaunchKernelGGL(ids_ss_kernel, dim3(TT * NN / 256), dim3(256), 0, stream,
                     Lbuf, Lmean, Lrstd, g_cn, b_cn, ne_h, Wce, bce, ids, ssb);
  hipLaunchKernelGGL(class_sum_kernel, dim3(NCC, TT), dim3(256), 0, stream, ids, ssb, ssum, cntb);
  hipLaunchKernelGGL(score_top_kernel, dim3(TT), dim3(32), 0, stream, ssum, cntb, Cs_out, cid_out, idxt);
  hipLaunchKernelGGL(mask_kernel, dim3(TT * NN / 256), dim3(256), 0, stream, ids, idxt, Tmask_out);

  // --- fused gram * state (WMMA) ---
  hipLaunchKernelGGL(gram_state_wmma, dim3(NN / 128, TT), dim3(256), 0, stream, ne_h, state, out1);

  // --- final MLPs ---
  hipLaunchKernelGGL(mlp1_kernel, dim3(H1S / 256, TT), dim3(256), 0, stream, out1, Wm1, bm1, out2);
  hipLaunchKernelGGL(mlp2_kernel, dim3(H2OUT / 256, TT), dim3(256), 0, stream, out2, Wm2, bm2, final_out);
}
